// CausalMultiHeadSelfAttention_62629213110280
// MI455X (gfx1250) — compile-verified, matrix operand order checked by reading, not run on hardware
//
#include <hip/hip_runtime.h>
#include <hip/hip_bf16.h>

typedef __attribute__((ext_vector_type(16))) __bf16 bf16x16;
typedef __attribute__((ext_vector_type(8)))  __bf16 bf16x8;
typedef __attribute__((ext_vector_type(4)))  __bf16 bf16x4;
typedef __attribute__((ext_vector_type(8)))  float  f32x8;
typedef __attribute__((ext_vector_type(4)))  float  f32x4;

#define Bb  2
#define Ss  2048
#define Dd  2048
#define Hh  16
#define DKk 128

// ---------------------------------------------------------------------------
// Fragment loader for CDNA5 bf16 WMMA A/B operands (wave32).
// Per ISA 7.12.2: lane L (r = L&15, kh = L>>4) holds two contiguous 8-element
// runs of the row at k = kh*8 and k = 16 + kh*8  ->  two 16B loads.
// ---------------------------------------------------------------------------
static __device__ __forceinline__ bf16x16 ldfrag(const __bf16* __restrict__ base, int ld) {
  const int lane = threadIdx.x & 31;
  const __bf16* p = base + (size_t)(lane & 15) * (size_t)ld + (lane >> 4) * 8;
  bf16x8 lo = *(const bf16x8*)p;
  bf16x8 hi = *(const bf16x8*)(p + 16);
  bf16x16 f;
#pragma unroll
  for (int i = 0; i < 8; ++i) { f[i] = lo[i]; f[i + 8] = hi[i]; }
  return f;
}

static __device__ __forceinline__ f32x8 wmma_bf16(bf16x16 a, bf16x16 b, f32x8 c) {
  return __builtin_amdgcn_wmma_f32_16x16x32_bf16(false, a, false, b, (short)0, c, false, false);
}

// ---------------------------------------------------------------------------
// fp32 -> bf16 convert, 4 elements / thread
// ---------------------------------------------------------------------------
__global__ __launch_bounds__(256) void cvt4_f32_bf16(const f32x4* __restrict__ s,
                                                     bf16x4* __restrict__ d, int n4) {
  int i = blockIdx.x * 256 + threadIdx.x;
  if (i < n4) {
    f32x4 v = s[i];
    bf16x4 o;
#pragma unroll
    for (int r = 0; r < 4; ++r) o[r] = (__bf16)v[r];
    d[i] = o;
  }
}

// ---------------------------------------------------------------------------
// GEMM  Y[M,N] = A[M,K] * W[N,K]^T  (both operands k-contiguous; bf16 in,
// f32 accumulate).  Block = 4 waves; wave tile = 32x64.
// MODE 0: bf16 out, row-major [M,N]
// MODE 1: f32  out, row-major [M,N]           (final projection -> d_out)
// MODE 2: bf16 out, transposed to [B,H,DK,S]  (V projection)
// ---------------------------------------------------------------------------
template <int MODE>
__global__ __launch_bounds__(128) void gemm_nt(const __bf16* __restrict__ A,
                                               const __bf16* __restrict__ W,
                                               void* __restrict__ Yv,
                                               int Kd, int N) {
  const int wave = threadIdx.x >> 5;
  const int lane = threadIdx.x & 31;
  const int n    = lane & 15;
  const int half = lane >> 4;
  const int row0 = blockIdx.y * 128 + wave * 32;
  const int col0 = blockIdx.x * 64;

  f32x8 acc[2][4] = {};
  const __bf16* a0p = A + (size_t)row0 * Kd;
  const __bf16* a1p = A + (size_t)(row0 + 16) * Kd;

  for (int k0 = 0; k0 < Kd; k0 += 32) {
    bf16x16 a0 = ldfrag(a0p + k0, Kd);
    bf16x16 a1 = ldfrag(a1p + k0, Kd);
#pragma unroll
    for (int j = 0; j < 4; ++j) {
      bf16x16 bf = ldfrag(W + (size_t)(col0 + 16 * j) * Kd + k0, Kd);
      acc[0][j] = wmma_bf16(a0, bf, acc[0][j]);
      acc[1][j] = wmma_bf16(a1, bf, acc[1][j]);
    }
  }

  if (MODE == 0) {
    __bf16* Y = (__bf16*)Yv;
#pragma unroll
    for (int i = 0; i < 2; ++i)
#pragma unroll
      for (int j = 0; j < 4; ++j)
#pragma unroll
        for (int r = 0; r < 8; ++r)
          Y[(size_t)(row0 + 16 * i + half * 8 + r) * N + (col0 + 16 * j + n)] =
              (__bf16)acc[i][j][r];
  } else if (MODE == 1) {
    float* Y = (float*)Yv;
#pragma unroll
    for (int i = 0; i < 2; ++i)
#pragma unroll
      for (int j = 0; j < 4; ++j)
#pragma unroll
        for (int r = 0; r < 8; ++r)
          Y[(size_t)(row0 + 16 * i + half * 8 + r) * N + (col0 + 16 * j + n)] =
              acc[i][j][r];
  } else {  // MODE 2: write V transposed [B][H][DK][S]; per-lane contiguous in s
    __bf16* Y = (__bf16*)Yv;
#pragma unroll
    for (int i = 0; i < 2; ++i) {
#pragma unroll
      for (int j = 0; j < 4; ++j) {
        const int col  = col0 + 16 * j + n;        // = h*128 + dk
        const int h    = col >> 7;
        const int dk   = col & 127;
        const int rowg = row0 + 16 * i + half * 8; // = b*S + s (8 consecutive s)
        const int b    = rowg >> 11;
        const int s    = rowg & 2047;
        bf16x8 w;
#pragma unroll
        for (int r = 0; r < 8; ++r) w[r] = (__bf16)acc[i][j][r];
        *(bf16x8*)(Y + (((size_t)b * Hh + h) * DKk + dk) * Ss + s) = w;
      }
    }
  }
}

// ---------------------------------------------------------------------------
// RoPE in place on bf16 [B,S,H,DK]; one thread per (even,odd) pair.
// ---------------------------------------------------------------------------
__global__ __launch_bounds__(256) void rope_bf16(__bf16* __restrict__ X,
                                                 const int* __restrict__ pos,
                                                 int total_pairs) {
  int i = blockIdx.x * 256 + threadIdx.x;
  if (i >= total_pairs) return;
  const int j    = i & 63;        // pair index within head (half = 64)
  const int rest = i >> 6;        // (b*S + s)*H + h
  const int s    = (rest >> 4) & (Ss - 1);
  const float p  = (float)pos[s];
  // inv_freq = 10000^(-j/64) = exp(-(j/64)*ln(10000))
  const float inv = __expf(-(float)j * (9.210340371976184f / 64.0f));
  float sn, cs;
  __sincosf(p * inv, &sn, &cs);
  __bf16* ptr = X + (size_t)rest * DKk + 2 * j;
  const float x1 = (float)ptr[0];
  const float x2 = (float)ptr[1];
  ptr[0] = (__bf16)(x1 * cs - x2 * sn);
  ptr[1] = (__bf16)(x1 * sn + x2 * cs);
}

// ---------------------------------------------------------------------------
// Flash attention: one wave = 16 queries x full DK=128 for one (b,h).
// Key chunks of 32: 8 score WMMAs + online softmax (shfl butterflies within
// 16-lane halves) + P bounced through padded LDS into A-fragment layout +
// 8 PV WMMAs against transposed V.
// ---------------------------------------------------------------------------
__global__ __launch_bounds__(128) void flash_attn(const __bf16* __restrict__ Q,
                                                  const __bf16* __restrict__ Kk,
                                                  const __bf16* __restrict__ Vt,
                                                  __bf16* __restrict__ O) {
  __shared__ __align__(16) __bf16 plds[4][16 * 40];  // 80B row pitch: conflict-free b128
  const int wave = threadIdx.x >> 5;
  const int lane = threadIdx.x & 31;
  const int n    = lane & 15;
  const int half = lane >> 4;
  const int wid  = blockIdx.x * 4 + wave;
  const int qt   = wid & 127;
  const int h    = (wid >> 7) & 15;
  const int b    = wid >> 11;
  const int q0   = qt * 16;
  __bf16* pl = plds[wave];

  const __bf16* Qbh = Q  + (size_t)b * Ss * Dd + (size_t)h * DKk;
  const __bf16* Kbh = Kk + (size_t)b * Ss * Dd + (size_t)h * DKk;
  const __bf16* Vbh = Vt + ((size_t)b * Hh + h) * (size_t)DKk * Ss;

  bf16x16 qf[4];
#pragma unroll
  for (int c = 0; c < 4; ++c) qf[c] = ldfrag(Qbh + (size_t)q0 * Dd + 32 * c, Dd);

  f32x8 of[8] = {};
  float rmax[8], rsum[8];
#pragma unroll
  for (int r = 0; r < 8; ++r) { rmax[r] = -1e30f; rsum[r] = 0.0f; }

  const float sc = 0.08838834764831845f;  // 1/sqrt(128)

  for (int k0 = 0; k0 < q0 + 16; k0 += 32) {
    f32x8 s0 = {}, s1 = {};
#pragma unroll
    for (int c = 0; c < 4; ++c) {
      bf16x16 kf = ldfrag(Kbh + (size_t)k0 * Dd + 32 * c, Dd);
      s0 = wmma_bf16(qf[c], kf, s0);
    }
#pragma unroll
    for (int c = 0; c < 4; ++c) {
      bf16x16 kf = ldfrag(Kbh + (size_t)(k0 + 16) * Dd + 32 * c, Dd);
      s1 = wmma_bf16(qf[c], kf, s1);
    }
    // online softmax over the 32-key chunk (C layout: row m = r + half*8, col n)
#pragma unroll
    for (int r = 0; r < 8; ++r) {
      const int q = q0 + r + half * 8;
      float a = s0[r] * sc;
      float e = s1[r] * sc;
      if (k0 + n > q)      a = -1e30f;
      if (k0 + 16 + n > q) e = -1e30f;
      float mx = fmaxf(a, e);
      mx = fmaxf(mx, __shfl_xor(mx, 1, 32));
      mx = fmaxf(mx, __shfl_xor(mx, 2, 32));
      mx = fmaxf(mx, __shfl_xor(mx, 4, 32));
      mx = fmaxf(mx, __shfl_xor(mx, 8, 32));
      const float nm   = fmaxf(rmax[r], mx);
      const float corr = __expf(rmax[r] - nm);
      rmax[r] = nm;
      const float p0 = __expf(a - nm);
      const float p1 = __expf(e - nm);
      float ps = p0 + p1;
      ps += __shfl_xor(ps, 1, 32);
      ps += __shfl_xor(ps, 2, 32);
      ps += __shfl_xor(ps, 4, 32);
      ps += __shfl_xor(ps, 8, 32);
      rsum[r] = rsum[r] * corr + ps;
#pragma unroll
      for (int j = 0; j < 8; ++j) of[j][r] *= corr;
      __bf16* row = pl + (r + half * 8) * 40;
      row[n]      = (__bf16)p0;
      row[n + 16] = (__bf16)p1;
    }
    // re-read P in A-fragment layout from LDS
    const __bf16* pr = pl + (lane & 15) * 40 + (lane >> 4) * 8;
    bf16x8 lo = *(const bf16x8*)pr;
    bf16x8 hi = *(const bf16x8*)(pr + 16);
    bf16x16 pf;
#pragma unroll
    for (int i = 0; i < 8; ++i) { pf[i] = lo[i]; pf[i + 8] = hi[i]; }
#pragma unroll
    for (int j = 0; j < 8; ++j) {
      bf16x16 vf = ldfrag(Vbh + (size_t)(16 * j) * Ss + k0, Ss);
      of[j] = wmma_bf16(pf, vf, of[j]);
    }
  }

#pragma unroll
  for (int r = 0; r < 8; ++r) rsum[r] = 1.0f / rsum[r];
#pragma unroll
  for (int j = 0; j < 8; ++j)
#pragma unroll
    for (int r = 0; r < 8; ++r)
      O[((size_t)b * Ss + (q0 + r + half * 8)) * Dd + (size_t)h * DKk + 16 * j + n] =
          (__bf16)(of[j][r] * rsum[r]);
}

// ---------------------------------------------------------------------------
extern "C" void kernel_launch(void* const* d_in, const int* in_sizes, int n_in,
                              void* d_out, int out_size, void* d_ws, size_t ws_size,
                              hipStream_t stream) {
  (void)in_sizes; (void)n_in; (void)out_size; (void)ws_size;
  const float* x  = (const float*)d_in[0];
  const float* wq = (const float*)d_in[1];
  const float* wk = (const float*)d_in[2];
  const float* wv = (const float*)d_in[3];
  const float* wo = (const float*)d_in[4];
  const int*  pos = (const int*)d_in[5];

  char* ws = (char*)d_ws;
  size_t off = 0;
  auto alloc = [&](size_t bytes) -> void* {
    void* p = ws + off;
    off += (bytes + 255) & ~(size_t)255;
    return p;
  };
  const size_t ND = (size_t)Bb * Ss * Dd;  // activation elements
  const size_t NW = (size_t)Dd * Dd;       // weight elements

  __bf16* xb  = (__bf16*)alloc(ND * 2);
  __bf16* wqb = (__bf16*)alloc(NW * 2);
  __bf16* wkb = (__bf16*)alloc(NW * 2);
  __bf16* wvb = (__bf16*)alloc(NW * 2);
  __bf16* wob = (__bf16*)alloc(NW * 2);
  __bf16* Qb  = (__bf16*)alloc(ND * 2);
  __bf16* Kb  = (__bf16*)alloc(ND * 2);
  __bf16* Vt  = (__bf16*)alloc(ND * 2);   // [B,H,DK,S]
  __bf16* Ob  = (__bf16*)alloc(ND * 2);

  // fp32 -> bf16
  {
    int n4 = (int)(ND / 4);
    cvt4_f32_bf16<<<(n4 + 255) / 256, 256, 0, stream>>>((const f32x4*)x, (bf16x4*)xb, n4);
    int w4 = (int)(NW / 4);
    dim3 g((w4 + 255) / 256);
    cvt4_f32_bf16<<<g, 256, 0, stream>>>((const f32x4*)wq, (bf16x4*)wqb, w4);
    cvt4_f32_bf16<<<g, 256, 0, stream>>>((const f32x4*)wk, (bf16x4*)wkb, w4);
    cvt4_f32_bf16<<<g, 256, 0, stream>>>((const f32x4*)wv, (bf16x4*)wvb, w4);
    cvt4_f32_bf16<<<g, 256, 0, stream>>>((const f32x4*)wo, (bf16x4*)wob, w4);
  }

  // Q/K/V projections (V written transposed)
  dim3 gg(Dd / 64, (Bb * Ss) / 128);
  gemm_nt<0><<<gg, 128, 0, stream>>>(xb, wqb, (void*)Qb, Dd, Dd);
  gemm_nt<0><<<gg, 128, 0, stream>>>(xb, wkb, (void*)Kb, Dd, Dd);
  gemm_nt<2><<<gg, 128, 0, stream>>>(xb, wvb, (void*)Vt, Dd, Dd);

  // RoPE on Q and K
  {
    int pairs = (int)(ND / 2);
    dim3 g((pairs + 255) / 256);
    rope_bf16<<<g, 256, 0, stream>>>(Qb, pos, pairs);
    rope_bf16<<<g, 256, 0, stream>>>(Kb, pos, pairs);
  }

  // flash attention: B*H*(S/16) = 4096 waves / 4 per block
  flash_attn<<<(Bb * Hh * (Ss / 16)) / 4, 128, 0, stream>>>(Qb, Kb, Vt, Ob);

  // output projection -> fp32 d_out
  gemm_nt<1><<<gg, 128, 0, stream>>>(Ob, wob, d_out, Dd, Dd);
}
